// GraphEncoder3D_84181359001796
// MI455X (gfx1250) — matrix operand order repaired and checked by reading, not compiled, not run on hardware
//
#include <hip/hip_runtime.h>
#include <hip/hip_bf16.h>

typedef __bf16 bf16_t;
typedef __attribute__((ext_vector_type(16))) __bf16 v16bf;
typedef __attribute__((ext_vector_type(8)))  __bf16 v8bf;
typedef __attribute__((ext_vector_type(8)))  float  v8f;

#define BATCH 1024
#define NKPS  24
#define NEDGE 552
#define NHID  128

// ---------- small helpers ----------
static __device__ __forceinline__ bf16_t f2bf(float f) {
  union { float f; unsigned u; } v; v.f = f;
  unsigned r = v.u + 0x7FFFu + ((v.u >> 16) & 1u);   // round-to-nearest-even
  unsigned short h = (unsigned short)(r >> 16);
  bf16_t out; __builtin_memcpy(&out, &h, 2); return out;
}
static __device__ __forceinline__ float bf2f(bf16_t b) {
  unsigned short h; __builtin_memcpy(&h, &b, 2);
  union { unsigned u; float f; } v; v.u = ((unsigned)h) << 16; return v.f;
}
static __device__ __forceinline__ float eluf(float x) {
  return x > 0.f ? x : (__expf(x) - 1.f);
}
static __device__ __forceinline__ v8f vzero8() {
  v8f z;
#pragma unroll
  for (int i = 0; i < 8; ++i) z[i] = 0.f;
  return z;
}

// A fragment 16x32 bf16 from LDS (row-major, leading dim ldk elements)
static __device__ __forceinline__ v16bf load_a_lds(const bf16_t* base, int ldk,
                                                   int rowBase, int k0, int lane) {
  int m = lane & 15, h = lane >> 4;
  const bf16_t* p = base + (size_t)(rowBase + m) * ldk + k0 + 16 * h;
  v8bf lo = *(const v8bf*)p;
  v8bf hi = *(const v8bf*)(p + 8);
  v16bf a;
#pragma unroll
  for (int i = 0; i < 8; ++i) { a[i] = lo[i]; a[i + 8] = hi[i]; }
  return a;
}

// B fragment = W^T tile (32x16) from row-major bf16 weights [H, ldk]
static __device__ __forceinline__ v16bf load_b_g(const bf16_t* __restrict__ w, int ldk,
                                                 int col0, int k0, int lane) {
  int n = lane & 15, kh = lane >> 4;
  return *(const v16bf*)(w + (size_t)(col0 + n) * ldk + k0 + 16 * kh);
}

static __device__ __forceinline__ v8f wmma_bf16(v16bf a, v16bf b, v8f c) {
  return __builtin_amdgcn_wmma_f32_16x16x32_bf16(false, a, false, b, (short)0, c,
                                                 false, false);
}

// ---------- prep kernels ----------
__global__ void zerof_kernel(float* p, int n) {
  int i = blockIdx.x * blockDim.x + threadIdx.x;
  if (i < n) p[i] = 0.f;
}

__global__ void cvt_w_kernel(const float* __restrict__ src, bf16_t* __restrict__ dst,
                             int H, int K, int KPAD) {
  for (int i = blockIdx.x * blockDim.x + threadIdx.x; i < H * KPAD;
       i += gridDim.x * blockDim.x) {
    int r = i / KPAD, c = i - r * KPAD;
    dst[i] = (c < K) ? f2bf(src[r * K + c]) : f2bf(0.f);
  }
}

__global__ void bn_finalize_kernel(const float* __restrict__ sum,
                                   const float* __restrict__ sumsq,
                                   const float* __restrict__ g,
                                   const float* __restrict__ be, float invM,
                                   float* __restrict__ scale,
                                   float* __restrict__ shift) {
  int c = threadIdx.x;  // 128 threads
  float mean = sum[c] * invM;
  float var  = sumsq[c] * invM - mean * mean;
  float s = g[c] * rsqrtf(var + 1e-5f);
  scale[c] = s;
  shift[c] = be[c] - mean * s;
}

// edge2node over BN2(h2): z[b,n,c] = sc[c]*(sum_j h2)/24 + sh[c]*23/24 (bf16 out)
__global__ void __launch_bounds__(128)
edge2node_kernel(const bf16_t* __restrict__ h2, const float* __restrict__ sc,
                 const float* __restrict__ sh, bf16_t* __restrict__ z) {
  int bn = blockIdx.x, c = threadIdx.x;
  int b = bn / NKPS, n = bn - b * NKPS;
  const bf16_t* base = h2 + ((size_t)b * NEDGE + (size_t)n * 23) * NHID + c;
  float s = 0.f;
#pragma unroll 23
  for (int j = 0; j < 23; ++j) s += bf2f(base[(size_t)j * NHID]);
  z[(size_t)bn * NHID + c] = f2bf(sc[c] * (s * (1.f / 24.f)) + sh[c] * (23.f / 24.f));
}

// ---------- fused 2-layer MLP with WMMA ----------
// Wave w owns output channels [w*16, w*16+16) and iterates all 8 row-tiles,
// so each weight fragment is loaded from global exactly once per block.
// MODE 0: A = inputs fp32 [M,6] (pad K to 32), no fold
// MODE 1: A = node2edge(BN1(h1)) -> K=256
// MODE 2: A = z (already folded)  -> K=128
// MODE 3: A = [node2edge(BN3(h3)), BN2(h2)] -> K=384
template <int MODE, int KPAD>
__global__ void __launch_bounds__(256)
mlp_kernel(const float* __restrict__ Afp, const bf16_t* __restrict__ Abf,
           const bf16_t* __restrict__ Abf2, const float* __restrict__ scA,
           const float* __restrict__ shA, const float* __restrict__ scA2,
           const float* __restrict__ shA2, const bf16_t* __restrict__ W1,
           const float* __restrict__ B1, const bf16_t* __restrict__ W2,
           const float* __restrict__ B2, bf16_t* __restrict__ Hout,
           float* __restrict__ gSum, float* __restrict__ gSumSq) {
  constexpr int AK = (KPAD < 128) ? 128 : KPAD;  // leading dim (A2/result overlay A)
  __shared__ __align__(64) bf16_t sA[128 * AK];
  __shared__ float sScale[KPAD];
  __shared__ float sShift[KPAD];
  __shared__ float sSum[128];
  __shared__ float sSumSq[128];

  const int tid = threadIdx.x;
  const int lane = tid & 31;
  const int wave = tid >> 5;
  const int gRow0 = blockIdx.x * 128;  // first global row of tile

  if (tid < 128) { sSum[tid] = 0.f; sSumSq[tid] = 0.f; }

  // resolve per-column BN scale/shift tables for the gather (MODE 1/3)
  if (MODE == 1 || MODE == 3) {
    for (int c = tid; c < KPAD; c += 256) {
      float sc, sh;
      if (c < 256) { sc = scA[c & 127]; sh = shA[c & 127]; }
      else         { sc = scA2[c - 256]; sh = shA2[c - 256]; }
      sScale[c] = sc; sShift[c] = sh;
    }
  }
  __syncthreads();

  // ---- stage A tile (128 x KPAD) into LDS in 8-element chunks ----
  {
    constexpr int CPR = KPAD / 8;  // chunks per row
    for (int cidx = tid; cidx < 128 * CPR; cidx += 256) {
      int r = cidx / CPR;
      int cc = (cidx - r * CPR) * 8;
      int grow = gRow0 + r;
      v8bf out;
      if (MODE == 0) {
#pragma unroll
        for (int i = 0; i < 8; ++i) {
          int col = cc + i;
          out[i] = f2bf(col < 6 ? Afp[(size_t)grow * 6 + col] : 0.f);
        }
      } else if (MODE == 2) {
        out = *(const v8bf*)(Abf + (size_t)grow * NHID + cc);
      } else {
        int b = grow / NEDGE, e = grow - b * NEDGE;
        int recv = e / 23, j = e - recv * 23;
        int send = j + (j >= recv ? 1 : 0);
        const bf16_t* src;
        int c0;
        if (cc < 128)       { src = Abf  + ((size_t)(b * NKPS + recv)) * NHID; c0 = cc; }
        else if (cc < 256)  { src = Abf  + ((size_t)(b * NKPS + send)) * NHID; c0 = cc - 128; }
        else                { src = Abf2 + (size_t)grow * NHID;                c0 = cc - 256; }
        v8bf in = *(const v8bf*)(src + c0);
#pragma unroll
        for (int i = 0; i < 8; ++i)
          out[i] = f2bf(sScale[cc + i] * bf2f(in[i]) + sShift[cc + i]);
      }
      *(v8bf*)(sA + (size_t)r * AK + cc) = out;
    }
  }
  __syncthreads();

  const int nn = lane & 15, nhalf = lane >> 4;
  const int ch = wave * 16 + nn;  // output channel owned by this lane

  // ---- layer 1: C1 = A @ W1^T (wave = column tile) ----
  v8f acc[8];
#pragma unroll
  for (int r = 0; r < 8; ++r) acc[r] = vzero8();

  for (int k0 = 0; k0 < KPAD; k0 += 32) {
    v16bf bf = load_b_g(W1, KPAD, wave * 16, k0, lane);
#pragma unroll
    for (int r = 0; r < 8; ++r) {
      v16bf a = load_a_lds(sA, AK, r * 16, k0, lane);
      acc[r] = wmma_bf16(a, bf, acc[r]);
    }
  }
  __syncthreads();  // all reads of A done before overwriting with A2

  // epilogue 1: bias + ELU -> bf16 A2 in LDS (overlay, stride AK)
  {
    float bias = B1[ch];
#pragma unroll
    for (int r = 0; r < 8; ++r) {
#pragma unroll
      for (int i = 0; i < 8; ++i) {
        int row = r * 16 + i + 8 * nhalf;
        sA[(size_t)row * AK + ch] = f2bf(eluf(acc[r][i] + bias));
      }
    }
  }
  __syncthreads();

  // ---- layer 2: C2 = A2 @ W2^T ----
#pragma unroll
  for (int r = 0; r < 8; ++r) acc[r] = vzero8();

  for (int k0 = 0; k0 < 128; k0 += 32) {
    v16bf bf = load_b_g(W2, 128, wave * 16, k0, lane);
#pragma unroll
    for (int r = 0; r < 8; ++r) {
      v16bf a = load_a_lds(sA, AK, r * 16, k0, lane);
      acc[r] = wmma_bf16(a, bf, acc[r]);
    }
  }
  __syncthreads();  // all reads of A2 done before overwriting with result

  // epilogue 2: bias + ELU -> packed bf16 result tile in LDS + channel stats
  {
    float bias = B2[ch];
    float psum = 0.f, psq = 0.f;
#pragma unroll
    for (int r = 0; r < 8; ++r) {
#pragma unroll
      for (int i = 0; i < 8; ++i) {
        int row = r * 16 + i + 8 * nhalf;
        float v = eluf(acc[r][i] + bias);
        sA[(size_t)row * 128 + ch] = f2bf(v);  // packed stride 128
        psum += v; psq += v * v;
      }
    }
    atomicAdd(&sSum[ch], psum);
    atomicAdd(&sSumSq[ch], psq);
  }
  __syncthreads();

  // coalesced b128 copy-out of the 128x128 bf16 tile
  {
    bf16_t* dst = Hout + (size_t)gRow0 * NHID;
    for (int c = tid; c < (128 * 128) / 8; c += 256) {
      *(v8bf*)(dst + (size_t)c * 8) = *(const v8bf*)(sA + (size_t)c * 8);
    }
  }
  if (tid < 128) {
    atomicAdd(&gSum[tid], sSum[tid]);
    atomicAdd(&gSumSq[tid], sSumSq[tid]);
  }
}

// ---------- final: out[b,:] = fo_w @ flatten(edge2node(BN4(h4))) + fo_b ----------
__global__ void __launch_bounds__(256)
final_kernel(const bf16_t* __restrict__ h4, const float* __restrict__ sc4,
             const float* __restrict__ sh4, const float* __restrict__ fow,
             const float* __restrict__ fob, float* __restrict__ out) {
  __shared__ float r0[256];
  __shared__ float r1[256];
  int b = blockIdx.x, tid = threadIdx.x;
  const bf16_t* hb = h4 + (size_t)b * NEDGE * NHID;
  float p0 = 0.f, p1 = 0.f;
  for (int idx = tid; idx < NEDGE * NHID; idx += 256) {
    int e = idx >> 7, c = idx & 127;
    int n = e / 23;
    float v = bf2f(hb[idx]) * sc4[c] * (1.f / 24.f);
    int nc = n * NHID + c;
    p0 += v * fow[nc];
    p1 += v * fow[NKPS * NHID + nc];
  }
  for (int idx = tid; idx < NKPS * NHID; idx += 256) {
    int c = idx & 127;
    float sv = sh4[c] * (23.f / 24.f);
    p0 += sv * fow[idx];
    p1 += sv * fow[NKPS * NHID + idx];
  }
  r0[tid] = p0; r1[tid] = p1;
  __syncthreads();
  for (int s = 128; s > 0; s >>= 1) {
    if (tid < s) { r0[tid] += r0[tid + s]; r1[tid] += r1[tid + s]; }
    __syncthreads();
  }
  if (tid == 0) {
    out[b * 2 + 0] = r0[0] + fob[0];
    out[b * 2 + 1] = r1[0] + fob[1];
  }
}

// ---------- host ----------
extern "C" void kernel_launch(void* const* d_in, const int* in_sizes, int n_in,
                              void* d_out, int out_size, void* d_ws, size_t ws_size,
                              hipStream_t stream) {
  (void)in_sizes; (void)n_in; (void)out_size; (void)ws_size;
  const float* inputs = (const float*)d_in[0];
  const float* fo_w = (const float*)d_in[3];
  const float* fo_b = (const float*)d_in[4];
  const float* mw1[4] = {(const float*)d_in[5],  (const float*)d_in[11],
                         (const float*)d_in[17], (const float*)d_in[23]};
  const float* mb1[4] = {(const float*)d_in[6],  (const float*)d_in[12],
                         (const float*)d_in[18], (const float*)d_in[24]};
  const float* mw2[4] = {(const float*)d_in[7],  (const float*)d_in[13],
                         (const float*)d_in[19], (const float*)d_in[25]};
  const float* mb2[4] = {(const float*)d_in[8],  (const float*)d_in[14],
                         (const float*)d_in[20], (const float*)d_in[26]};
  const float* mg[4]  = {(const float*)d_in[9],  (const float*)d_in[15],
                         (const float*)d_in[21], (const float*)d_in[27]};
  const float* mbe[4] = {(const float*)d_in[10], (const float*)d_in[16],
                         (const float*)d_in[22], (const float*)d_in[28]};

  const size_t M_node = (size_t)BATCH * NKPS;   // 24576
  const size_t M_edge = (size_t)BATCH * NEDGE;  // 565248

  // workspace bump allocator
  char* base = (char*)d_ws;
  size_t off = 0;
  auto alloc = [&](size_t bytes) {
    void* p = base + off;
    off = (off + bytes + 255) & ~(size_t)255;
    return p;
  };
  float* stats = (float*)alloc(1024 * sizeof(float));       // 4 stages x (sum,sumsq)
  float* scsh  = (float*)alloc(1024 * sizeof(float));       // 4 stages x (scale,shift)
  const int KP1 = 32, KP2 = 256, KP3 = 128, KP4 = 384;
  bf16_t* wb1a = (bf16_t*)alloc((size_t)NHID * KP1 * 2);
  bf16_t* wb1b = (bf16_t*)alloc((size_t)NHID * 128 * 2);
  bf16_t* wb2a = (bf16_t*)alloc((size_t)NHID * KP2 * 2);
  bf16_t* wb2b = (bf16_t*)alloc((size_t)NHID * 128 * 2);
  bf16_t* wb3a = (bf16_t*)alloc((size_t)NHID * KP3 * 2);
  bf16_t* wb3b = (bf16_t*)alloc((size_t)NHID * 128 * 2);
  bf16_t* wb4a = (bf16_t*)alloc((size_t)NHID * KP4 * 2);
  bf16_t* wb4b = (bf16_t*)alloc((size_t)NHID * 128 * 2);
  bf16_t* h1 = (bf16_t*)alloc(M_node * NHID * 2);
  bf16_t* h2 = (bf16_t*)alloc(M_edge * NHID * 2);
  bf16_t* z3 = (bf16_t*)alloc(M_node * NHID * 2);
  bf16_t* h3 = (bf16_t*)alloc(M_node * NHID * 2);
  bf16_t* h4 = (bf16_t*)alloc(M_edge * NHID * 2);

  float* sum1 = stats + 0,   *ss1 = stats + 128;
  float* sum2 = stats + 256, *ss2 = stats + 384;
  float* sum3 = stats + 512, *ss3 = stats + 640;
  float* sum4 = stats + 768, *ss4 = stats + 896;
  float* sc1 = scsh + 0,   *sh1 = scsh + 128;
  float* sc2 = scsh + 256, *sh2 = scsh + 384;
  float* sc3 = scsh + 512, *sh3 = scsh + 640;
  float* sc4 = scsh + 768, *sh4 = scsh + 896;

  // zero stats; convert weights to bf16 (K padded to multiple of 32)
  zerof_kernel<<<4, 256, 0, stream>>>(stats, 1024);
  cvt_w_kernel<<<8,   256, 0, stream>>>(mw1[0], wb1a, NHID, 6,   KP1);
  cvt_w_kernel<<<64,  256, 0, stream>>>(mw2[0], wb1b, NHID, 128, 128);
  cvt_w_kernel<<<128, 256, 0, stream>>>(mw1[1], wb2a, NHID, 256, KP2);
  cvt_w_kernel<<<64,  256, 0, stream>>>(mw2[1], wb2b, NHID, 128, 128);
  cvt_w_kernel<<<64,  256, 0, stream>>>(mw1[2], wb3a, NHID, 128, KP3);
  cvt_w_kernel<<<64,  256, 0, stream>>>(mw2[2], wb3b, NHID, 128, 128);
  cvt_w_kernel<<<192, 256, 0, stream>>>(mw1[3], wb4a, NHID, 384, KP4);
  cvt_w_kernel<<<64,  256, 0, stream>>>(mw2[3], wb4b, NHID, 128, 128);

  // m1
  mlp_kernel<0, KP1><<<(int)(M_node / 128), 256, 0, stream>>>(
      inputs, nullptr, nullptr, nullptr, nullptr, nullptr, nullptr,
      wb1a, mb1[0], wb1b, mb2[0], h1, sum1, ss1);
  bn_finalize_kernel<<<1, 128, 0, stream>>>(sum1, ss1, mg[0], mbe[0],
                                            1.f / (float)M_node, sc1, sh1);
  // m2 (node2edge gather of BN1(h1))
  mlp_kernel<1, KP2><<<(int)(M_edge / 128), 256, 0, stream>>>(
      nullptr, h1, nullptr, sc1, sh1, nullptr, nullptr,
      wb2a, mb1[1], wb2b, mb2[1], h2, sum2, ss2);
  bn_finalize_kernel<<<1, 128, 0, stream>>>(sum2, ss2, mg[1], mbe[1],
                                            1.f / (float)M_edge, sc2, sh2);
  // edge2node(BN2(h2)) -> z3
  edge2node_kernel<<<(int)M_node, 128, 0, stream>>>(h2, sc2, sh2, z3);
  // m3
  mlp_kernel<2, KP3><<<(int)(M_node / 128), 256, 0, stream>>>(
      nullptr, z3, nullptr, nullptr, nullptr, nullptr, nullptr,
      wb3a, mb1[2], wb3b, mb2[2], h3, sum3, ss3);
  bn_finalize_kernel<<<1, 128, 0, stream>>>(sum3, ss3, mg[2], mbe[2],
                                            1.f / (float)M_node, sc3, sh3);
  // m4 ([node2edge(BN3(h3)), BN2(h2)])
  mlp_kernel<3, KP4><<<(int)(M_edge / 128), 256, 0, stream>>>(
      nullptr, h3, h2, sc3, sh3, sc2, sh2,
      wb4a, mb1[3], wb4b, mb2[3], h4, sum4, ss4);
  bn_finalize_kernel<<<1, 128, 0, stream>>>(sum4, ss4, mg[3], mbe[3],
                                            1.f / (float)M_edge, sc4, sh4);
  // edge2node(BN4(h4)) -> flatten -> fo_w
  final_kernel<<<BATCH, 256, 0, stream>>>(h4, sc4, sh4, fo_w, fo_b, (float*)d_out);
}
